// GamestateEncoder_34059090657444
// MI455X (gfx1250) — matrix operand order repaired and checked by reading, compiled, not run on hardware
//
#include <hip/hip_runtime.h>
#include <hip/hip_bf16.h>
#include <stdint.h>

typedef __attribute__((ext_vector_type(16))) _Float16 v16h;
typedef __attribute__((ext_vector_type(8)))  float    v8f;

#define NTOK   8192     // 16*512
#define HDIM   256
#define SEQ    512
#define NB     16
#define NHEADS 8
#define HD     32
#define COMBK  2048
#define FFK    1024
#define EPSLN  1e-5f
#define ATTN_SCALE 0.17677669529663687f   // 1/sqrt(32)

// ---------------------------------------------------------------------------
// feature -> (group, slot) tables.  groups: 0=cont 1=cat 2=bool 3=count 4=time 5=angle
#define G(g,p) ((unsigned short)(((g)<<8)|(p)))
__device__ __constant__ unsigned short FEAT_TAB[128] = {
  G(0,0),G(0,1),G(1,0),G(2,0),G(1,1),G(1,2),G(1,3),G(1,4),G(4,0),G(0,2),G(0,3),G(0,4),G(5,0),G(5,1),
  G(1,5),G(1,6),G(1,7),G(1,8),G(1,9),G(1,10),G(1,11),G(1,12),G(1,13),G(1,14),G(1,15),G(1,16),G(1,17),
  G(1,18),G(1,19),G(1,20),G(1,21),G(1,22),G(1,23),G(1,24),G(1,25),G(1,26),G(1,27),G(1,28),G(1,29),
  G(1,30),G(1,31),G(1,32),
  G(2,1),G(2,2),G(3,0),G(1,33),G(0,5),G(0,6),G(2,3),G(3,1),G(1,34),G(0,7),G(0,8),G(2,4),G(3,2),G(1,35),
  G(0,9),G(0,10),G(2,5),G(3,3),G(1,36),G(0,11),G(0,12),G(1,59),G(4,1),G(4,2),G(3,4),G(1,37),G(0,13),
  G(0,14),G(1,38),G(0,15),G(0,16),G(1,39),G(0,17),G(0,18),G(1,40),G(0,19),G(0,20),G(1,41),G(0,21),
  G(0,22),G(1,42),G(0,23),G(0,24),G(1,43),G(0,25),G(0,26),G(1,44),G(0,27),G(0,28),G(1,45),G(0,29),
  G(0,30),G(1,46),G(0,31),G(0,32),G(1,47),G(0,33),G(0,34),G(1,48),G(0,35),G(0,36),G(1,49),G(0,37),
  G(0,38),G(1,50),G(0,39),G(0,40),G(1,51),G(0,41),G(0,42),G(1,52),G(0,43),G(0,44),G(1,53),G(0,45),
  G(0,46),G(1,54),G(0,47),G(0,48),G(1,55),G(0,49),G(0,50),G(1,56),G(1,57),G(1,58),G(4,3)
};
__device__ __constant__ int GROUP_OFF[6] = {0, 816, 1776, 1872, 1952, 2016};

// ---------------------------------------------------------------------------
__device__ inline v8f wmma_f16(v16h a, v16h b, v8f c) {
  return __builtin_amdgcn_wmma_f32_16x16x32_f16(false, a, false, b, (short)0, c, false, false);
}

// A-tile lane swizzle (16x32 f16 A fragment, contiguous k, same as B):
//   lane = r + 16*(k>>4),      j = k & 15
// B-tile lane order (32x16 f16 B fragment):
//   lane = (k>>4)*16 + n,      j = k & 15

template<int OUTMODE>
__device__ inline void store_out(void* outp, int m, int n, int N, float v) {
  if constexpr (OUTMODE == 0) {                 // plain f16 [M][N]
    ((_Float16*)outp)[(size_t)m * N + n] = (_Float16)v;
  } else if constexpr (OUTMODE == 3) {          // plain f32 [M][N]
    ((float*)outp)[(size_t)m * N + n] = v;
  } else {
    int bb = m >> 9, s = m & 511, head = n >> 5, d = n & 31;
    int bh = bb * NHEADS + head;
    if constexpr (OUTMODE == 1) {               // q: packed A-tiles, pre-scaled
      int mt = s >> 4, r = s & 15;
      int lane = r + (d >> 4) * 16;
      int j = d & 15;
      ((_Float16*)outp)[(((size_t)bh * 32 + mt) * 32 + lane) * 16 + j] =
          (_Float16)(v * ATTN_SCALE);
    } else if constexpr (OUTMODE == 2) {        // kT: packed B-tiles (scores GEMM)
      int nt = s >> 4;
      int lane = (d >> 4) * 16 + (s & 15);
      int j = d & 15;
      ((_Float16*)outp)[(((size_t)bh * 32 + nt) * 32 + lane) * 16 + j] = (_Float16)v;
    } else {                                    // OUTMODE 4  v: packed B-tiles (ctx GEMM)
      int kt = s >> 5;
      int lane = ((s >> 4) & 1) * 16 + (d & 15);
      int j = s & 15;
      ((_Float16*)outp)[((((size_t)bh * 16 + kt) * 2 + (d >> 4)) * 32 + lane) * 16 + j] =
          (_Float16)v;
    }
  }
}

// ---------------------------------------------------------------------------
// WMMA GEMM: C[M,N] = A[M,K](f16, row-major) * Bp(f16, packed tiles) + bias
// block = 256 threads (8 waves); 32x256 output tile; grid = (M/32, N/256)
// each wave: 2 M-tiles x 2 N-tiles = 4 WMMAs per 32-wide k-step
template<bool LNORM, bool RELU, bool RESID, int OUTMODE>
__global__ __launch_bounds__(256) void gemm32_kernel(
    const _Float16* __restrict__ A, int lda,
    const _Float16* __restrict__ Bp,
    const float* __restrict__ bias,
    const _Float16* __restrict__ res,
    const float* __restrict__ lng, const float* __restrict__ lnb,
    void* __restrict__ outp, int M, int N, int K)
{
  __shared__ alignas(32) _Float16 As[2][2][512];   // [buf][mt][lane*16+j]
  __shared__ float stage[LNORM ? 32 * 256 : 1];
  __shared__ float smean[32], srstd[32];

  int tid = threadIdx.x;
  int w = tid >> 5, lane = tid & 31, l16 = lane & 15, h16 = lane >> 4;
  int m0  = blockIdx.x * 32;
  int nb0 = blockIdx.y * 256;
  int nbase = nb0 + w * 32;
  int ntiles = N >> 4;
  int nt0 = nbase >> 4;

  v8f acc[2][2] = {};
  int nk = K >> 5;
  for (int it = 0; it < nk; ++it) {
    int k0 = it << 5;
    // cooperative, pre-swizzled A-tile staging: one dword per thread x2
#pragma unroll
    for (int t = 0; t < 2; ++t) {
      int hi = (tid + t * 256) * 2;          // half index 0..1022
      int r = hi >> 5, c = hi & 31;          // r 0..31, c even
      uint32_t dv = *(const uint32_t*)(A + (size_t)(m0 + r) * lda + k0 + c);
      int lane_dst = (r & 15) + (c >> 4) * 16;
      int j = c & 15;
      *(uint32_t*)&As[it & 1][r >> 4][lane_dst * 16 + j] = dv;
    }
    __syncthreads();
    v16h a0 = *(const v16h*)&As[it & 1][0][lane * 16];
    v16h a1 = *(const v16h*)&As[it & 1][1][lane * 16];
    const _Float16* bb = Bp + (((size_t)it * ntiles + nt0) * 32 + lane) * 16;
    v16h b0 = *(const v16h*)bb;
    v16h b1 = *(const v16h*)(bb + 512);      // next n-tile
    acc[0][0] = wmma_f16(a0, b0, acc[0][0]);
    acc[0][1] = wmma_f16(a0, b1, acc[0][1]);
    acc[1][0] = wmma_f16(a1, b0, acc[1][0]);
    acc[1][1] = wmma_f16(a1, b1, acc[1][1]);
  }

  float bv0 = bias[nbase + l16];
  float bv1 = bias[nbase + 16 + l16];

  if (!LNORM) {
#pragma unroll
    for (int mt = 0; mt < 2; ++mt)
#pragma unroll
      for (int i = 0; i < 8; ++i) {
        int m = m0 + mt * 16 + i + h16 * 8;
        float v0 = acc[mt][0][i] + bv0;
        float v1 = acc[mt][1][i] + bv1;
        if (RESID) {
          v0 += (float)res[(size_t)m * N + nbase + l16];
          v1 += (float)res[(size_t)m * N + nbase + 16 + l16];
        }
        if (RELU) { v0 = fmaxf(v0, 0.f); v1 = fmaxf(v1, 0.f); }
        store_out<OUTMODE>(outp, m, nbase + l16, N, v0);
        store_out<OUTMODE>(outp, m, nbase + 16 + l16, N, v1);
      }
  } else {
    // fused row LayerNorm (N == 256, grid.y == 1)
#pragma unroll
    for (int mt = 0; mt < 2; ++mt)
#pragma unroll
      for (int i = 0; i < 8; ++i) {
        int r = mt * 16 + i + h16 * 8;
        float v0 = acc[mt][0][i] + bv0;
        float v1 = acc[mt][1][i] + bv1;
        if (RESID) {
          int m = m0 + r;
          v0 += (float)res[(size_t)m * N + nbase + l16];
          v1 += (float)res[(size_t)m * N + nbase + 16 + l16];
        }
        stage[r * 256 + w * 32 + l16]      = v0;
        stage[r * 256 + w * 32 + 16 + l16] = v1;
      }
    __syncthreads();
    {   // stats: 8 threads per row, shuffle-reduce within 8-lane group
      int r = tid >> 3, c0 = tid & 7;
      float s = 0.f, ss = 0.f;
#pragma unroll
      for (int j = 0; j < 32; ++j) {
        float v = stage[r * 256 + c0 + j * 8];
        s += v; ss += v * v;
      }
      for (int msk = 4; msk >= 1; msk >>= 1) { s += __shfl_xor(s, msk); ss += __shfl_xor(ss, msk); }
      if (c0 == 0) {
        float mean = s * (1.f / 256.f);
        float var  = ss * (1.f / 256.f) - mean * mean;
        smean[r] = mean;
        srstd[r] = rsqrtf(var + EPSLN);
      }
    }
    __syncthreads();
    {
      int r = tid >> 3, c0 = tid & 7;
      float mean = smean[r], rstd = srstd[r];
      int m = m0 + r;
#pragma unroll
      for (int j = 0; j < 32; ++j) {
        int c = c0 + j * 8;
        float v = (stage[r * 256 + c] - mean) * rstd * lng[c] + lnb[c];
        if (RELU) v = fmaxf(v, 0.f);
        store_out<OUTMODE>(outp, m, nb0 + c, N, v);
      }
    }
  }
}

// ---------------------------------------------------------------------------
// attention core: grid = (SEQ/16, B*NHEADS), block = 256 (8 waves)
__global__ __launch_bounds__(256) void attn_core_kernel(
    const _Float16* __restrict__ qp,   // packed A-tiles, pre-scaled: [bh][mt32][lane][16]
    const _Float16* __restrict__ kp,   // packed B-tiles: [bh][nt32][lane][16]
    const _Float16* __restrict__ vp,   // packed B-tiles: [bh][kt16][nt2][lane][16]
    _Float16* __restrict__ ctx)        // [token, 256] = [b,s, h*32+d]
{
  __shared__ float sc[16 * 512];                    // scores f32, reused as partials
  __shared__ alignas(32) _Float16 wt[16 * 512];     // probs in packed A-tile order

  int tid = threadIdx.x;
  int w = tid >> 5, lane = tid & 31, l16 = lane & 15, h16 = lane >> 4;
  int mt0 = blockIdx.x;
  int bh = blockIdx.y;

  // ---- scores: q(16x32) x kT(32x512); K=32 in one WMMA; 8 waves x 4 N-tiles
  v16h a = *(const v16h*)(qp + (((size_t)bh * 32 + mt0) * 32 + lane) * 16);
#pragma unroll
  for (int j = 0; j < 4; ++j) {
    int nt = w * 4 + j;
    v16h b = *(const v16h*)(kp + (((size_t)bh * 32 + nt) * 32 + lane) * 16);
    v8f c = {};
    c = wmma_f16(a, b, c);
#pragma unroll
    for (int i = 0; i < 8; ++i)
      sc[(i + h16 * 8) * 512 + nt * 16 + l16] = c[i];
  }
  __syncthreads();

  // ---- softmax: wave w handles rows 2w, 2w+1; write probs packed for A loads
#pragma unroll
  for (int rr = 0; rr < 2; ++rr) {
    int r = w * 2 + rr;
    float mx = -1e30f;
    for (int c = lane; c < 512; c += 32) mx = fmaxf(mx, sc[r * 512 + c]);
    for (int msk = 16; msk >= 1; msk >>= 1) mx = fmaxf(mx, __shfl_xor(mx, msk));
    float s = 0.f;
    for (int c = lane; c < 512; c += 32) {
      float e = __expf(sc[r * 512 + c] - mx);
      sc[r * 512 + c] = e;
      s += e;
    }
    for (int msk = 16; msk >= 1; msk >>= 1) s += __shfl_xor(s, msk);
    float inv = 1.f / s;
    for (int c = lane; c < 512; c += 32) {
      int kt = c >> 5, kl = c & 31;
      int lane_dst = r + (kl >> 4) * 16;
      int j = kl & 15;
      wt[kt * 512 + lane_dst * 16 + j] = (_Float16)(sc[r * 512 + c] * inv);
    }
  }
  __syncthreads();

  // ---- ctx = probs(16x512) x v(512x32); K split across waves (64 each)
  v8f c0 = {}, c1 = {};
#pragma unroll
  for (int kk = 0; kk < 2; ++kk) {
    int kt = w * 2 + kk;
    v16h a2 = *(const v16h*)&wt[kt * 512 + lane * 16];
    const _Float16* vb = vp + ((size_t)bh * 16 + kt) * 1024 + lane * 16;
    v16h b0 = *(const v16h*)vb;
    v16h b1 = *(const v16h*)(vb + 512);
    c0 = wmma_f16(a2, b0, c0);
    c1 = wmma_f16(a2, b1, c1);
  }
  float* part = sc;   // reuse: [8][16][32]
#pragma unroll
  for (int i = 0; i < 8; ++i) {
    int r = i + h16 * 8;
    part[(w * 16 + r) * 32 + l16]      = c0[i];
    part[(w * 16 + r) * 32 + 16 + l16] = c1[i];
  }
  __syncthreads();
  for (int idx = tid; idx < 512; idx += 256) {
    int r = idx >> 5, c = idx & 31;
    float s = 0.f;
#pragma unroll
    for (int ww = 0; ww < 8; ++ww) s += part[(ww * 16 + r) * 32 + c];
    int bb = bh >> 3, head = bh & 7;
    ctx[((size_t)(bb * SEQ + mt0 * 16 + r)) * HDIM + head * HD + c] = (_Float16)s;
  }
}

// ---------------------------------------------------------------------------
// feature encoder: one block per token, 128 threads (one per feature)
struct EncPtrs {   // scalar groups: 0=cont 1=bool 2=count 3=time 4=angle
  const float* w[5];
  const float* b[5];
  const float* g[5];
  const float* bn[5];
};

__global__ __launch_bounds__(128) void encode_kernel(
    const float* __restrict__ x,
    const float* __restrict__ pre_g, const float* __restrict__ pre_b,
    EncPtrs ep,
    const float* __restrict__ emb, const float* __restrict__ cat_g,
    const float* __restrict__ cat_bn,
    _Float16* __restrict__ comb)
{
  __shared__ float red[128];
  __shared__ float s_mean, s_rstd;
  int tok = blockIdx.x;
  int f = threadIdx.x;
  float xv = x[(size_t)tok * 128 + f];

  red[f] = xv;
  __syncthreads();
  for (int s = 64; s > 0; s >>= 1) { if (f < s) red[f] += red[f + s]; __syncthreads(); }
  if (f == 0) s_mean = red[0] * (1.f / 128.f);
  __syncthreads();
  float d = xv - s_mean;
  red[f] = d * d;
  __syncthreads();
  for (int s = 64; s > 0; s >>= 1) { if (f < s) red[f] += red[f + s]; __syncthreads(); }
  if (f == 0) s_rstd = rsqrtf(red[0] * (1.f / 128.f) + EPSLN);
  __syncthreads();

  float xp = d * s_rstd * pre_g[f] + pre_b[f];

  int tab = FEAT_TAB[f];
  int g = tab >> 8, pos = tab & 255;
  float vals[16];
  const float* lg;
  const float* lb;
  if (g == 1) {   // categorical: embedding gather
    int id = (int)xp;
    id = id < 0 ? 0 : (id > 9999 ? 9999 : id);
    const float* e = emb + (size_t)id * 16;
#pragma unroll
    for (int j = 0; j < 16; ++j) vals[j] = e[j];
    lg = cat_g; lb = cat_bn;
  } else {        // scalar: Linear(1,16)
    int si = (g == 0) ? 0 : g - 1;
    const float* w = ep.w[si];
    const float* b = ep.b[si];
#pragma unroll
    for (int j = 0; j < 16; ++j) vals[j] = xp * w[j] + b[j];
    lg = ep.g[si]; lb = ep.bn[si];
  }
  float m = 0.f;
#pragma unroll
  for (int j = 0; j < 16; ++j) m += vals[j];
  m *= (1.f / 16.f);
  float vv = 0.f;
#pragma unroll
  for (int j = 0; j < 16; ++j) { float t = vals[j] - m; vv += t * t; }
  float rs = rsqrtf(vv * (1.f / 16.f) + EPSLN);
  size_t base = (size_t)tok * COMBK + GROUP_OFF[g] + pos * 16;
#pragma unroll
  for (int j = 0; j < 16; ++j) {
    float y = (vals[j] - m) * rs * lg[j] + lb[j];
    comb[base + j] = (_Float16)fmaxf(y, 0.f);
  }
}

// ---------------------------------------------------------------------------
// fused f32->f16 conversion + WMMA B-tile packing of a [K,N] weight matrix
__global__ void pack_b_kernel(const float* __restrict__ src,
                              _Float16* __restrict__ dst, int K, int N) {
  int idx = blockIdx.x * 256 + threadIdx.x;
  if (idx >= K * N) return;
  int j = idx & 15;
  int lane = (idx >> 4) & 31;
  int t = idx >> 9;
  int ntiles = N >> 4;
  int kt = t / ntiles, nt = t - kt * ntiles;
  int k = kt * 32 + (lane >> 4) * 16 + j;
  int n = nt * 16 + (lane & 15);
  dst[idx] = (_Float16)src[(size_t)k * N + n];
}

// ---------------------------------------------------------------------------
extern "C" void kernel_launch(void* const* d_in, const int* in_sizes, int n_in,
                              void* d_out, int out_size, void* d_ws, size_t ws_size,
                              hipStream_t stream) {
  (void)in_sizes; (void)n_in; (void)out_size; (void)ws_size;

  const float* x       = (const float*)d_in[0];
  const float* pre_g   = (const float*)d_in[1];
  const float* pre_b   = (const float*)d_in[2];
  EncPtrs ep;
  for (int s = 0; s < 5; ++s) {       // cont, bool, count, time, angle
    ep.w[s]  = (const float*)d_in[3 + 4 * s + 0];
    ep.b[s]  = (const float*)d_in[3 + 4 * s + 1];
    ep.g[s]  = (const float*)d_in[3 + 4 * s + 2];
    ep.bn[s] = (const float*)d_in[3 + 4 * s + 3];
  }
  const float* emb     = (const float*)d_in[23];
  const float* cat_g   = (const float*)d_in[24];
  const float* cat_bn  = (const float*)d_in[25];
  const float* comb_w  = (const float*)d_in[26];
  const float* comb_b  = (const float*)d_in[27];
  const float* comb_g  = (const float*)d_in[28];
  const float* comb_bn = (const float*)d_in[29];
  const float* ffn_w1  = (const float*)d_in[46];
  const float* ffn_b1  = (const float*)d_in[47];
  const float* ffn_w2  = (const float*)d_in[48];
  const float* ffn_b2  = (const float*)d_in[49];
  const float* n1_g    = (const float*)d_in[50];
  const float* n1_b    = (const float*)d_in[51];
  const float* n2_g    = (const float*)d_in[52];
  const float* n2_b    = (const float*)d_in[53];

  // ---- workspace carve-up (all f16, 32B-aligned chunks)
  _Float16* ws = (_Float16*)d_ws;
  size_t cur = 0;
  auto alloc = [&](size_t n) { _Float16* p = ws + cur; cur += n; return p; };
  _Float16* comb    = alloc((size_t)NTOK * COMBK);
  _Float16* h       = alloc((size_t)NTOK * HDIM);
  _Float16* h2      = alloc((size_t)NTOK * HDIM);
  _Float16* qbuf    = alloc((size_t)NTOK * HDIM);
  _Float16* kTbuf   = alloc((size_t)NTOK * HDIM);
  _Float16* vbuf    = alloc((size_t)NTOK * HDIM);
  _Float16* ctx     = alloc((size_t)NTOK * HDIM);
  _Float16* ffm     = alloc((size_t)NTOK * FFK);
  _Float16* combw16 = alloc((size_t)COMBK * HDIM);
  _Float16* aw16[2][4];
  for (int l = 0; l < 2; ++l)
    for (int mp = 0; mp < 4; ++mp)
      aw16[l][mp] = alloc((size_t)HDIM * HDIM);
  _Float16* w1_16 = alloc((size_t)HDIM * FFK);
  _Float16* w2_16 = alloc((size_t)FFK * HDIM);

  auto pack = [&](const float* src, _Float16* dst, int K, int N) {
    pack_b_kernel<<<(K * N + 255) / 256, 256, 0, stream>>>(src, dst, K, N);
  };
  pack(comb_w, combw16, COMBK, HDIM);
  for (int l = 0; l < 2; ++l)
    for (int mp = 0; mp < 4; ++mp)
      pack((const float*)d_in[30 + 8 * l + 2 * mp], aw16[l][mp], HDIM, HDIM);
  pack(ffn_w1, w1_16, HDIM, FFK);
  pack(ffn_w2, w2_16, FFK, HDIM);

  // ---- feature encoding -> comb [8192, 2048] f16
  encode_kernel<<<NTOK, 128, 0, stream>>>(x, pre_g, pre_b, ep, emb, cat_g, cat_bn, comb);

  // ---- h = relu(ln(comb @ W + b))
  gemm32_kernel<true, true, false, 0><<<dim3(NTOK / 32, 1), 256, 0, stream>>>(
      comb, COMBK, combw16, comb_b, nullptr, comb_g, comb_bn, h, NTOK, HDIM, COMBK);

  // ---- attention blocks
  auto attn = [&](const _Float16* hin, _Float16* hout, int layer) {
    const float* qb = (const float*)d_in[31 + 8 * layer];
    const float* kb = (const float*)d_in[33 + 8 * layer];
    const float* vb = (const float*)d_in[35 + 8 * layer];
    const float* ob = (const float*)d_in[37 + 8 * layer];
    gemm32_kernel<false, false, false, 1><<<dim3(NTOK / 32, 1), 256, 0, stream>>>(
        hin, HDIM, aw16[layer][0], qb, nullptr, nullptr, nullptr, qbuf, NTOK, HDIM, HDIM);
    gemm32_kernel<false, false, false, 2><<<dim3(NTOK / 32, 1), 256, 0, stream>>>(
        hin, HDIM, aw16[layer][1], kb, nullptr, nullptr, nullptr, kTbuf, NTOK, HDIM, HDIM);
    gemm32_kernel<false, false, false, 4><<<dim3(NTOK / 32, 1), 256, 0, stream>>>(
        hin, HDIM, aw16[layer][2], vb, nullptr, nullptr, nullptr, vbuf, NTOK, HDIM, HDIM);
    attn_core_kernel<<<dim3(SEQ / 16, NB * NHEADS), 256, 0, stream>>>(qbuf, kTbuf, vbuf, ctx);
    gemm32_kernel<true, false, true, 0><<<dim3(NTOK / 32, 1), 256, 0, stream>>>(
        ctx, HDIM, aw16[layer][3], ob, hin, n1_g, n1_b, hout, NTOK, HDIM, HDIM);
  };
  attn(h, h2, 0);    // h2 = ln(h + attn1(h))
  attn(h2, h, 1);    // h  = ln(h2 + attn2(h2))

  // ---- FFN
  gemm32_kernel<false, true, false, 0><<<dim3(NTOK / 32, FFK / 256), 256, 0, stream>>>(
      h, HDIM, w1_16, ffn_b1, nullptr, nullptr, nullptr, ffm, NTOK, FFK, HDIM);
  gemm32_kernel<true, false, true, 3><<<dim3(NTOK / 32, 1), 256, 0, stream>>>(
      ffm, FFK, w2_16, ffn_b2, h, n2_g, n2_b, d_out, NTOK, HDIM, FFK);
}
